// FunnelTransformer_67894843015328
// MI455X (gfx1250) — matrix operand order repaired and checked by reading, not compiled, not run on hardware
//
#include <hip/hip_runtime.h>

// ---------------------------------------------------------------------------
// Funnel-transformer encoder forward for MI455X (gfx1250).
// ~258 GFLOP of GEMM on ~140MB resident state (fits 192MB L2) -> matrix-bound.
// All GEMM work via v_wmma_f32_16x16x32_bf16. A/K tiles staged to LDS by the
// Tensor Data Mover (tensor_load_to_lds + s_wait_tensorcnt) with TDM LDS
// padding producing conflict-free strides; double-buffered so TDM + global
// loads overlap the WMMA chain.
// ---------------------------------------------------------------------------

typedef __attribute__((ext_vector_type(16))) __bf16 bf16x16;
typedef __attribute__((ext_vector_type(8)))  __bf16 bf16x8;
typedef __attribute__((ext_vector_type(8)))  float  f32x8;
typedef __attribute__((ext_vector_type(4)))  unsigned u32x4;
typedef __attribute__((ext_vector_type(8)))  int      i32x8;
typedef __attribute__((ext_vector_type(4)))  int      i32x4;

constexpr int B_ = 16, S_ = 512, D_ = 512, H_ = 8, DK_ = 64, DFF_ = 1024, L_ = 6;

#define USE_TDM 1
#if __has_include(<hip/amd_detail/amd_gfx1250_TDM.h>)
#define TDM_SIX_ARG 1
#else
#define TDM_SIX_ARG 0
#endif

#define CAT16(lo, hi) __builtin_shufflevector(lo, hi, 0,1,2,3,4,5,6,7,8,9,10,11,12,13,14,15)

#if defined(__HIP_DEVICE_COMPILE__)
// Issue a 2D TDM tile load (bf16 elements) into LDS, with LDS padding.
// pad_interval: threshold = (1<<pi) 8-byte units;  pad_amount: (pa+1) DWORDs.
__device__ __forceinline__ void tdm_load_2d(unsigned lds_addr, const void* gptr,
                                            unsigned tensor_w, unsigned tensor_h,
                                            unsigned tile_w, unsigned tile_h,
                                            unsigned stride_elems,
                                            unsigned pad_interval, unsigned pad_amount) {
  unsigned long long ga = (unsigned long long)gptr;
  u32x4 g0 = { 1u,                                   // count=1, user descriptor
               lds_addr,                             // lds_addr (bytes)
               (unsigned)ga,                         // global_addr[31:0]
               (unsigned)(ga >> 32) | (2u << 30) };  // global_addr[56:32] | type=2
  unsigned w0 = (1u << 16)                           // data_size = 2 bytes
              | (1u << 20)                           // pad_enable
              | (pad_interval << 22) | (pad_amount << 25);
  i32x8 g1 = { (int)w0,
               (int)(tensor_w << 16),                            // tensor_dim0 lo16
               (int)((tensor_w >> 16) | (tensor_h << 16)),       // td0 hi | td1 lo
               (int)((tensor_h >> 16) | (tile_w << 16)),         // td1 hi | tile_dim0
               (int)tile_h,                                      // tile_dim1 | tile_dim2=0
               (int)stride_elems,                                // td0_stride[31:0]
               0, 0 };
  i32x4 z4 = { 0, 0, 0, 0 };
#if TDM_SIX_ARG
  i32x8 z8 = { 0, 0, 0, 0, 0, 0, 0, 0 };
  __builtin_amdgcn_tensor_load_to_lds(g0, g1, z4, z4, z8, 0);
#else
  __builtin_amdgcn_tensor_load_to_lds(g0, g1, z4, z4, 0);
#endif
}
#endif

// ------------------------------ embedding ----------------------------------
__global__ void k_embed(const int* __restrict__ seq, const int* __restrict__ pos,
                        const float* __restrict__ we, const float* __restrict__ pe,
                        float* __restrict__ x) {
  int row = blockIdx.x;               // b*S + s
  int t = threadIdx.x;                // 128 threads, 4 floats each
  int tok = seq[row], p = pos[row];
  const float4* w4 = (const float4*)(we + (size_t)tok * D_);
  const float4* p4 = (const float4*)(pe + (size_t)p * D_);
  float4* o4 = (float4*)(x + (size_t)row * D_);
  float4 a = w4[t], c = p4[t];
  o4[t] = make_float4(a.x + c.x, a.y + c.y, a.z + c.z, a.w + c.w);
}

// ------------------------------ f32 -> bf16 --------------------------------
__global__ void k_cvt(const float* __restrict__ in, __bf16* __restrict__ out, int n) {
  for (int i = blockIdx.x * blockDim.x + threadIdx.x; i < n; i += gridDim.x * blockDim.x)
    out[i] = (__bf16)in[i];
}

// --------------------- skip probability + bernoulli ------------------------
__global__ void k_skip(const float* __restrict__ x, const float* __restrict__ sw,
                       const float* __restrict__ sb, int* __restrict__ keep,
                       unsigned seed) {
  int lane = threadIdx.x & 31, wv = threadIdx.x >> 5;
  int row = blockIdx.x * 8 + wv;      // b*S + s
  const float* xr = x + (size_t)row * D_;
  float acc = 0.f;
  for (int d = lane; d < D_; d += 32) acc += xr[d] * sw[d];
  for (int m = 16; m; m >>= 1) acc += __shfl_xor(acc, m, 32);
  if (lane == 0) {
    float p = 1.f / (1.f + __expf(-(acc + sb[0])));
    unsigned h = (unsigned)row * 0x9E3779B1u ^ (seed + 0x7F4A7C15u);
    h ^= h >> 16; h *= 0x85EBCA6Bu; h ^= h >> 13; h *= 0xC2B2AE35u; h ^= h >> 16;
    float u = (float)(h >> 8) * (1.0f / 16777216.0f);
    keep[row] = (u < p) ? 0 : 1;      // drop ~ Bernoulli(p); keep = !drop
  }
}

// ----------------- stable order (kept first) via LDS scan ------------------
__global__ void k_order(const int* __restrict__ keep, int* __restrict__ srcidx,
                        int* __restrict__ nk) {
  __shared__ int s[S_];
  int b = blockIdx.x, t = threadIdx.x;
  int f = keep[b * S_ + t];
  s[t] = f;
  __syncthreads();
  for (int off = 1; off < S_; off <<= 1) {
    int v = (t >= off) ? s[t - off] : 0;
    __syncthreads();
    s[t] += v;
    __syncthreads();
  }
  int incl = s[t], excl = incl - f, total = s[S_ - 1];
  int dest = f ? excl : (total + (t - excl));
  srcidx[b * S_ + dest] = t;
  if (t == 0) nk[b] = total;
}

// -------------------- gather/compact + bf16 + pad mask ---------------------
__global__ void k_gather(const float* __restrict__ xin, float* __restrict__ xout,
                         __bf16* __restrict__ xb, unsigned char* __restrict__ pad,
                         const int* __restrict__ srcidx, const int* __restrict__ nk) {
  int row = blockIdx.x;               // b*S + i (dest)
  int b = row >> 9, i = row & 511;
  int src = srcidx[row];
  bool kd = i < nk[b];
  const float4* in4 = (const float4*)(xin + ((size_t)((b << 9) | src)) * D_);
  float4* out4 = (float4*)(xout + (size_t)row * D_);
  __bf16* ob = xb + (size_t)row * D_;
  int t = threadIdx.x;                // 128 threads
  float4 v = kd ? in4[t] : make_float4(0.f, 0.f, 0.f, 0.f);
  out4[t] = v;
  ob[4 * t + 0] = (__bf16)v.x; ob[4 * t + 1] = (__bf16)v.y;
  ob[4 * t + 2] = (__bf16)v.z; ob[4 * t + 3] = (__bf16)v.w;
  if (t == 0) pad[row] = kd ? 0 : 1;
}

// ------------------------- generic batched WMMA GEMM -----------------------
// C[z] (MxN,f32 or bf16) = A[b] (MxK) @ W[h] (KxN) (+bias)(+relu)
// z in grid.z; b = z / nH; h = z % nH. M,K,N multiples of 128/32/64.
// Double-buffered: TDM(A) + global(B) for step k+1 overlap WMMAs of step k.
__global__ __launch_bounds__(256)
void k_gemm(const __bf16* __restrict__ A, const __bf16* __restrict__ W,
            const float* __restrict__ bias, float* __restrict__ C,
            __bf16* __restrict__ Cb, int M, int N, int K, int nH, int relu) {
#if defined(__HIP_DEVICE_COMPILE__)
  __shared__ __bf16 Al[2][128][40];   // padded stride 40: conflict-free b128 reads
  __shared__ __bf16 Bt[2][64][40];    // B tile transposed [n][k]
  int z = blockIdx.z, bidx = z / nH, h = z % nH;
  A += (size_t)bidx * M * K;
  W += (size_t)h * K * N;
  size_t cofs = (size_t)z * M * N;
  int tm = blockIdx.y * 128, tn = blockIdx.x * 64;
  int t = threadIdx.x, lane = t & 31, wv = t >> 5;
  f32x8 acc[4] = {};
  int mrow = wv * 16 + (lane & 15);
  int akb = (lane < 16) ? 0 : 8;
  int bkb = (lane < 16) ? 0 : 16;
  int ncol = lane & 15;
  unsigned ldsA0 = (unsigned)(unsigned long long)(void*)&Al[0][0][0];
  unsigned ldsA1 = (unsigned)(unsigned long long)(void*)&Al[1][0][0];

  auto stage_a = [&](int k0, int buf) {
#if USE_TDM
    // A tile 128x32 via TDM (wave 0 issues; TDM ignores EXEC). Row = 32 elems
    // = 64B -> pad_interval=3; pad 4 DWORDs (pad_amount=3) -> stride 40 elems.
    if (wv == 0)
      tdm_load_2d(buf ? ldsA1 : ldsA0, &A[(size_t)tm * K + k0], (unsigned)K,
                  (unsigned)M, 32u, 128u, (unsigned)K, 3u, 3u);
#else
    #pragma unroll
    for (int c = 0; c < 2; ++c) {
      int ci = t + c * 256;
      int r = ci >> 2, cc = (ci & 3) * 8;
      *(bf16x8*)&Al[buf][r][cc] = *(const bf16x8*)&A[(size_t)(tm + r) * K + k0 + cc];
    }
#endif
  };
  auto stage_b = [&](int k0, int buf) {
    unsigned bv[4];
    #pragma unroll
    for (int c = 0; c < 4; ++c) {     // loads first (pipelined) ...
      int pi = t + c * 256;
      int kk = pi >> 5, np = (pi & 31) * 2;
      bv[c] = *(const unsigned*)&W[(size_t)(k0 + kk) * N + tn + np];
    }
    #pragma unroll
    for (int c = 0; c < 4; ++c) {     // ... then transposed stores
      int pi = t + c * 256;
      int kk = pi >> 5, np = (pi & 31) * 2;
      Bt[buf][np][kk]     = __builtin_bit_cast(__bf16, (unsigned short)(bv[c] & 0xffffu));
      Bt[buf][np + 1][kk] = __builtin_bit_cast(__bf16, (unsigned short)(bv[c] >> 16));
    }
  };

  // prologue: stage step 0
  stage_a(0, 0);
  stage_b(0, 0);
#if USE_TDM
  if (wv == 0) __builtin_amdgcn_s_wait_tensorcnt(0);
#endif
  __syncthreads();

  int nsteps = K >> 5;
  for (int ki = 0; ki < nsteps; ++ki) {
    int cur = ki & 1, nxt = cur ^ 1;
    int k0 = ki << 5;
    if (ki + 1 < nsteps) {            // stage k+1 into other buffer (no wait)
      stage_a(k0 + 32, nxt);
      stage_b(k0 + 32, nxt);
      if (k0 + 64 < K)                // warm L2 two tiles ahead
        __builtin_prefetch(&W[(size_t)(k0 + 64 + (t & 31)) * N + tn], 0, 1);
    }
    bf16x8 alo = *(const bf16x8*)&Al[cur][mrow][akb];
    bf16x8 ahi = *(const bf16x8*)&Al[cur][mrow][akb + 16];
    bf16x16 af = CAT16(alo, ahi);
    #pragma unroll
    for (int nt = 0; nt < 4; ++nt) {
      bf16x8 blo = *(const bf16x8*)&Bt[cur][nt * 16 + ncol][akb];
      bf16x8 bhi = *(const bf16x8*)&Bt[cur][nt * 16 + ncol][akb + 16];
      bf16x16 bfv = CAT16(blo, bhi);
      acc[nt] = __builtin_amdgcn_wmma_f32_16x16x32_bf16(
          false, af, false, bfv, (short)0, acc[nt], false, false);
    }
#if USE_TDM
    if (wv == 0 && ki + 1 < nsteps) __builtin_amdgcn_s_wait_tensorcnt(0);
#endif
    __syncthreads();
  }
  #pragma unroll
  for (int nt = 0; nt < 4; ++nt) {
    int col = tn + nt * 16 + ncol;
    float bs = bias ? bias[col] : 0.f;
    #pragma unroll
    for (int r = 0; r < 8; ++r) {
      int row = tm + wv * 16 + r + ((lane < 16) ? 0 : 8);
      float v = acc[nt][r] + bs;
      if (relu) v = fmaxf(v, 0.f);
      if (Cb) Cb[cofs + (size_t)row * N + col] = (__bf16)v;
      else    C [cofs + (size_t)row * N + col] = v;
    }
  }
#endif
}

// -------------------------- flash attention (WMMA) -------------------------
// Q,K,V: [B,H,S,64] bf16. O: [B,S,H*64] f32. pad: [B,S] bytes (1 = masked key)
__global__ __launch_bounds__(128)
void k_attn(const __bf16* __restrict__ Q, const __bf16* __restrict__ Km,
            const __bf16* __restrict__ V, const unsigned char* __restrict__ pad,
            float* __restrict__ O, float scale) {
#if defined(__HIP_DEVICE_COMPILE__)
  __shared__ __bf16 Kl[2][32][72];    // 32 keys x 64 dk (padded stride 72)
  __shared__ __bf16 Vt[2][64][40];    // transposed: [dv][key]
  __shared__ __bf16 Pl[4][16][40];    // per-wave probs, [qrow][key]
  int b = blockIdx.z, h = blockIdx.y;
  int t = threadIdx.x, lane = t & 31, wv = t >> 5;
  size_t base = ((size_t)(b * H_ + h)) * S_ * DK_;
  int q0 = blockIdx.x * 64 + wv * 16;
  int m16 = lane & 15;
  int half8 = (lane < 16) ? 0 : 8;
  int akb = (lane < 16) ? 0 : 8;
  int bkb = (lane < 16) ? 0 : 16;
  unsigned ldsK0 = (unsigned)(unsigned long long)(void*)&Kl[0][0][0];
  unsigned ldsK1 = (unsigned)(unsigned long long)(void*)&Kl[1][0][0];
  // Q fragments over kdim {0..31},{32..63}
  const __bf16* qrow = Q + base + (size_t)(q0 + m16) * DK_;
  bf16x16 qf0 = CAT16(*(const bf16x8*)&qrow[akb], *(const bf16x8*)&qrow[akb + 16]);
  bf16x16 qf1 = CAT16(*(const bf16x8*)&qrow[32 + akb], *(const bf16x8*)&qrow[32 + akb + 16]);

  f32x8 acc[4] = {};
  float mrow[8], lrow[8];
  #pragma unroll
  for (int r = 0; r < 8; ++r) { mrow[r] = -3.0e38f; lrow[r] = 0.f; }

  auto stage_k = [&](int kt, int buf) {
#if USE_TDM
    // K tile 32x64 via TDM: row = 64 elems = 128B -> pad_interval=4;
    // pad 4 DWORDs (pad_amount=3) -> LDS row stride 72 elems.
    if (wv == 0)
      tdm_load_2d(buf ? ldsK1 : ldsK0, &Km[base + (size_t)kt * DK_],
                  (unsigned)DK_, (unsigned)S_, 64u, 32u, (unsigned)DK_, 4u, 3u);
#else
    #pragma unroll
    for (int c = 0; c < 2; ++c) {
      int ci = t + c * 128;
      int kr = ci >> 3, cc = (ci & 7) * 8;
      *(bf16x8*)&Kl[buf][kr][cc] = *(const bf16x8*)&Km[base + (size_t)(kt + kr) * DK_ + cc];
    }
#endif
  };
  auto stage_v = [&](int kt, int buf) {
    unsigned vv[8];
    #pragma unroll
    for (int c = 0; c < 8; ++c) {
      int pi = t + c * 128;
      int kr = pi >> 5, dv = (pi & 31) * 2;
      vv[c] = *(const unsigned*)&V[base + (size_t)(kt + kr) * DK_ + dv];
    }
    #pragma unroll
    for (int c = 0; c < 8; ++c) {
      int pi = t + c * 128;
      int kr = pi >> 5, dv = (pi & 31) * 2;
      Vt[buf][dv][kr]     = __builtin_bit_cast(__bf16, (unsigned short)(vv[c] & 0xffffu));
      Vt[buf][dv + 1][kr] = __builtin_bit_cast(__bf16, (unsigned short)(vv[c] >> 16));
    }
  };

  stage_k(0, 0);
  stage_v(0, 0);
#if USE_TDM
  if (wv == 0) __builtin_amdgcn_s_wait_tensorcnt(0);
#endif
  __syncthreads();

  for (int kt = 0; kt < S_; kt += 32) {
    int cur = (kt >> 5) & 1, nxt = cur ^ 1;
    if (kt + 32 < S_) {               // stage next key tile (no wait yet)
      stage_k(kt + 32, nxt);
      stage_v(kt + 32, nxt);
    }
    // scores S = Q @ K^T : 2 key subtiles x 2 kdim steps
    f32x8 c0 = {}, c1 = {};
    #pragma unroll
    for (int ks = 0; ks < 2; ++ks) {
      bf16x16 aq = ks ? qf1 : qf0;
      int kofs = ks * 32 + akb;
      {
        const __bf16* kp = &Kl[cur][m16][kofs];
        bf16x16 bfv = CAT16(*(const bf16x8*)kp, *(const bf16x8*)(kp + 16));
        c0 = __builtin_amdgcn_wmma_f32_16x16x32_bf16(false, aq, false, bfv, (short)0, c0, false, false);
      }
      {
        const __bf16* kp = &Kl[cur][16 + m16][kofs];
        bf16x16 bfv = CAT16(*(const bf16x8*)kp, *(const bf16x8*)(kp + 16));
        c1 = __builtin_amdgcn_wmma_f32_16x16x32_bf16(false, aq, false, bfv, (short)0, c1, false, false);
      }
    }
    // scale + pad mask + online softmax
    bool p0m = pad[b * S_ + kt + m16] != 0;
    bool p1m = pad[b * S_ + kt + 16 + m16] != 0;
    float s0[8], s1[8], nm[8];
    #pragma unroll
    for (int r = 0; r < 8; ++r) {
      s0[r] = p0m ? -1e9f : c0[r] * scale;
      s1[r] = p1m ? -1e9f : c1[r] * scale;
      float v = fmaxf(s0[r], s1[r]);
      v = fmaxf(v, __shfl_xor(v, 1, 32)); v = fmaxf(v, __shfl_xor(v, 2, 32));
      v = fmaxf(v, __shfl_xor(v, 4, 32)); v = fmaxf(v, __shfl_xor(v, 8, 32));
      nm[r] = fmaxf(mrow[r], v);
    }
    #pragma unroll
    for (int r = 0; r < 8; ++r) {
      float p0 = __expf(s0[r] - nm[r]);
      float p1 = __expf(s1[r] - nm[r]);
      float rs = p0 + p1;
      rs += __shfl_xor(rs, 1, 32); rs += __shfl_xor(rs, 2, 32);
      rs += __shfl_xor(rs, 4, 32); rs += __shfl_xor(rs, 8, 32);
      float alpha = __expf(mrow[r] - nm[r]);
      lrow[r] = lrow[r] * alpha + rs;
      mrow[r] = nm[r];
      #pragma unroll
      for (int nt = 0; nt < 4; ++nt) acc[nt][r] *= alpha;
      int prow = r + half8;
      Pl[wv][prow][m16]      = (__bf16)p0;
      Pl[wv][prow][m16 + 16] = (__bf16)p1;
    }
    // O += P @ V  (K-dim = 32 keys; Pl is wave-private -> in-order LDS, no barrier)
    {
      const __bf16* pp = &Pl[wv][m16][akb];
      bf16x16 pf = CAT16(*(const bf16x8*)pp, *(const bf16x8*)(pp + 16));
      #pragma unroll
      for (int nt = 0; nt < 4; ++nt) {
        const __bf16* vp = &Vt[cur][nt * 16 + m16][akb];
        bf16x16 vf = CAT16(*(const bf16x8*)vp, *(const bf16x8*)(vp + 16));
        acc[nt] = __builtin_amdgcn_wmma_f32_16x16x32_bf16(
            false, pf, false, vf, (short)0, acc[nt], false, false);
      }
    }
#if USE_TDM
    if (wv == 0 && kt + 32 < S_) __builtin_amdgcn_s_wait_tensorcnt(0);
#endif
    __syncthreads();
  }
  size_t obase = (size_t)(b * S_) * (H_ * DK_) + h * DK_;
  #pragma unroll
  for (int r = 0; r < 8; ++r) {
    int qr = q0 + r + half8;
    float inv = 1.f / lrow[r];
    #pragma unroll
    for (int nt = 0; nt < 4; ++nt)
      O[obase + (size_t)qr * (H_ * DK_) + nt * 16 + m16] = acc[nt][r] * inv;
  }
#endif
}

// ------------------- residual add + layernorm (ddof=1) ---------------------
__global__ void k_ln(const float* __restrict__ z1, const float* __restrict__ z2,
                     const float* __restrict__ g, const float* __restrict__ bta,
                     float* __restrict__ out, __bf16* __restrict__ outb) {
  int lane = threadIdx.x & 31, wv = threadIdx.x >> 5;
  int row = blockIdx.x * 8 + wv;
  const float* a = z1 + (size_t)row * D_;
  const float* c = z2 + (size_t)row * D_;
  float v[16], s = 0.f, ss = 0.f;
  #pragma unroll
  for (int i = 0; i < 16; ++i) {
    int d = lane + 32 * i;
    float x = a[d] + c[d];
    v[i] = x; s += x; ss += x * x;
  }
  for (int m = 16; m; m >>= 1) { s += __shfl_xor(s, m, 32); ss += __shfl_xor(ss, m, 32); }
  float mu = s * (1.f / D_);
  float var = (ss - (float)D_ * mu * mu) * (1.f / (D_ - 1));
  float inv = 1.f / (sqrtf(fmaxf(var, 0.f)) + 1e-3f);
  #pragma unroll
  for (int i = 0; i < 16; ++i) {
    int d = lane + 32 * i;
    float y = g[d] * (v[i] - mu) * inv + bta[d];
    out[(size_t)row * D_ + d] = y;
    if (outb) outb[(size_t)row * D_ + d] = (__bf16)y;
  }
}

// ------------------------------- launcher ----------------------------------
extern "C" void kernel_launch(void* const* d_in, const int* in_sizes, int n_in,
                              void* d_out, int out_size, void* d_ws, size_t ws_size,
                              hipStream_t stream) {
  (void)in_sizes; (void)n_in; (void)out_size; (void)ws_size;
  const int*   seq  = (const int*)d_in[0];
  const int*   pos  = (const int*)d_in[1];
  const float* we   = (const float*)d_in[2];
  const float* pe   = (const float*)d_in[3];
  const float* Wq   = (const float*)d_in[4];
  const float* Wk   = (const float*)d_in[5];
  const float* Wv   = (const float*)d_in[6];
  const float* Wo   = (const float*)d_in[7];
  const float* bo   = (const float*)d_in[8];
  const float* ln1a = (const float*)d_in[9];
  const float* ln1b = (const float*)d_in[10];
  const float* w1   = (const float*)d_in[11];
  const float* b1   = (const float*)d_in[12];
  const float* w2   = (const float*)d_in[13];
  const float* b2   = (const float*)d_in[14];
  const float* ln2a = (const float*)d_in[15];
  const float* ln2b = (const float*)d_in[16];
  const float* skw  = (const float*)d_in[17];
  const float* skb  = (const float*)d_in[18];

  char* ws = (char*)d_ws;
  size_t off = 0;
  auto alloc = [&](size_t bytes) -> void* {
    void* p = ws + off;
    off = (off + bytes + 255) & ~(size_t)255;
    return p;
  };
  const size_t XF = (size_t)B_ * S_ * D_;          // 4,194,304
  const size_t HF = (size_t)B_ * S_ * DFF_;        // 8,388,608
  float*  xA  = (float*)alloc(XF * 4);             // persistent activation / attn-out
  float*  xB  = (float*)alloc(XF * 4);             // compacted x (residual 1)
  float*  xln = (float*)alloc(XF * 4);             // LN1 out (residual 2)
  __bf16* abf = (__bf16*)alloc(XF * 2);            // bf16 activation (QKV / FFN1 input)
  __bf16* obf = (__bf16*)alloc(XF * 2);            // bf16 attention-out
  __bf16* qb  = (__bf16*)alloc(XF * 2);
  __bf16* kb  = (__bf16*)alloc(XF * 2);
  __bf16* vb  = (__bf16*)alloc(XF * 2);
  float*  hf  = (float*)alloc(HF * 4);             // FFN hidden f32
  float*  yf  = hf;                                // aliased: GEMM f32 out (safe ordering)
  __bf16* hb  = (__bf16*)alloc(HF * 2);
  __bf16* wqb = (__bf16*)alloc((size_t)H_ * D_ * DK_ * 2);
  __bf16* wkb = (__bf16*)alloc((size_t)H_ * D_ * DK_ * 2);
  __bf16* wvb = (__bf16*)alloc((size_t)H_ * D_ * DK_ * 2);
  __bf16* wob = (__bf16*)alloc((size_t)D_ * D_ * 2);
  __bf16* w1b = (__bf16*)alloc((size_t)D_ * DFF_ * 2);
  __bf16* w2b = (__bf16*)alloc((size_t)DFF_ * D_ * 2);
  int* keep = (int*)alloc(B_ * S_ * 4);
  int* srci = (int*)alloc(B_ * S_ * 4);
  int* nk   = (int*)alloc(B_ * 4);
  unsigned char* pad = (unsigned char*)alloc(B_ * S_);

  const float scale = 0.044194173824159216f;       // 1/sqrt(512)

  k_embed<<<B_ * S_, 128, 0, stream>>>(seq, pos, we, pe, xA);

  for (int l = 0; l < L_; ++l) {
    // layer weights -> bf16
    k_cvt<<<512, 256, 0, stream>>>(Wq + (size_t)l * H_ * D_ * DK_, wqb, H_ * D_ * DK_);
    k_cvt<<<512, 256, 0, stream>>>(Wk + (size_t)l * H_ * D_ * DK_, wkb, H_ * D_ * DK_);
    k_cvt<<<512, 256, 0, stream>>>(Wv + (size_t)l * H_ * D_ * DK_, wvb, H_ * D_ * DK_);
    k_cvt<<<512, 256, 0, stream>>>(Wo + (size_t)l * D_ * D_,   wob, D_ * D_);
    k_cvt<<<512, 256, 0, stream>>>(w1 + (size_t)l * D_ * DFF_, w1b, D_ * DFF_);
    k_cvt<<<512, 256, 0, stream>>>(w2 + (size_t)l * DFF_ * D_, w2b, DFF_ * D_);
    // probabilistic skipper + stable compaction
    k_skip<<<B_ * S_ / 8, 256, 0, stream>>>(xA, skw + (size_t)l * D_, skb + l, keep,
                                            0x5851F42Du + 0x9E3779B1u * (unsigned)l);
    k_order<<<B_, 512, 0, stream>>>(keep, srci, nk);
    k_gather<<<B_ * S_, 128, 0, stream>>>(xA, xB, abf, pad, srci, nk);
    // QKV projections: batched over (b,h), M=512 N=64 K=512
    dim3 gqkv(1, 4, B_ * H_);
    k_gemm<<<gqkv, 256, 0, stream>>>(abf, wqb, nullptr, nullptr, qb, S_, DK_, D_, H_, 0);
    k_gemm<<<gqkv, 256, 0, stream>>>(abf, wkb, nullptr, nullptr, kb, S_, DK_, D_, H_, 0);
    k_gemm<<<gqkv, 256, 0, stream>>>(abf, wvb, nullptr, nullptr, vb, S_, DK_, D_, H_, 0);
    // flash attention -> xA (old x buffer is dead after gather)
    k_attn<<<dim3(S_ / 64, H_, B_), 128, 0, stream>>>(qb, kb, vb, pad, xA, scale);
    k_cvt<<<2048, 256, 0, stream>>>(xA, obf, (int)XF);
    // output projection + bias
    k_gemm<<<dim3(8, 4, B_), 256, 0, stream>>>(obf, wob, bo + (size_t)l * D_,
                                               yf, nullptr, S_, D_, D_, 1, 0);
    // LN1(attn_out + residual) -> xln (f32) + abf (bf16)
    k_ln<<<B_ * S_ / 8, 256, 0, stream>>>(yf, xB, ln1a + (size_t)l * D_,
                                          ln1b + (size_t)l * D_, xln, abf);
    // FFN
    k_gemm<<<dim3(16, 4, B_), 256, 0, stream>>>(abf, w1b, b1 + (size_t)l * DFF_,
                                                hf, nullptr, S_, DFF_, D_, 1, 1);
    k_cvt<<<2048, 256, 0, stream>>>(hf, hb, (int)HF);
    k_gemm<<<dim3(8, 4, B_), 256, 0, stream>>>(hb, w2b, b2 + (size_t)l * D_,
                                               yf, nullptr, S_, D_, DFF_, 1, 0);
    // LN2(ffn_out + xln) -> xA for next layer
    k_ln<<<B_ * S_ / 8, 256, 0, stream>>>(yf, xln, ln2a + (size_t)l * D_,
                                          ln2b + (size_t)l * D_, xA, nullptr);
  }
  hipMemcpyAsync(d_out, xA, XF * sizeof(float), hipMemcpyDeviceToDevice, stream);
}
